// GroupedQueryAttention_31327491457065
// MI455X (gfx1250) — compile-verified, matrix operand order checked by reading, not run on hardware
//
#include <hip/hip_runtime.h>
#include <hip/hip_bf16.h>

typedef __attribute__((ext_vector_type(16))) __bf16       v16bf;
typedef __attribute__((ext_vector_type(8)))  float        v8f;
typedef __attribute__((ext_vector_type(4)))  unsigned int u32x4;
typedef __attribute__((ext_vector_type(8)))  int          i32x8;
typedef __attribute__((ext_vector_type(4)))  int          i32x4;

#define B_     32
#define S_     16
#define DIM_   2048
#define H_     8
#define QH_    32
#define DH_    64
#define CACHE_ 2048
#define TOTAL_ 2048
#define G_     4
#define KT     32   // keys per attention tile

#if defined(__has_builtin)
#if __has_builtin(__builtin_amdgcn_tensor_load_to_lds) && __has_builtin(__builtin_amdgcn_s_wait_tensorcnt)
#define HAVE_TDM 1
#endif
#endif
#ifndef HAVE_TDM
#define HAVE_TDM 0
#endif

// fp32 -> bf16 via native conversion (v_cvt_pk_bf16_f32 class ops on gfx1250)
static __device__ __forceinline__ __bf16 f2bf(float f) { return (__bf16)f; }

// K-index pattern for 16-bit A/B WMMA fragments (ISA 7.12.2):
// VGPR v<4 holds K = 2v,2v+1 (+8 for upper half-wave); v>=4 holds K = 16+2(v-4)... (+8)
static __device__ __forceinline__ int kpat(int v, int hv) {
  return (v < 4 ? 2 * v : 16 + 2 * (v - 4)) + 8 * hv;
}

#if HAVE_TDM
// Issue one TDM DMA: contiguous 1-D tile of `nelem` fp32 global -> LDS.
// D# group0: count=1 | lds_addr | global_addr(57b) | type=2 ("image").
// D# group1: data_size=4B, tensor_dim0 = tile_dim0 = nelem, tensor_dim1=1,
//            tensor_dim0_stride = nelem, no padding / iterate / multicast.
static __device__ __forceinline__ void tdm_load_1d(unsigned lds_addr,
                                                   unsigned long long gaddr,
                                                   unsigned nelem) {
  u32x4 g0 = {1u,                                   // count=1, user descriptor
              lds_addr,                             // bits 63:32
              (unsigned)(gaddr & 0xFFFFFFFFu),      // bits 95:64
              (unsigned)((gaddr >> 32) & 0x01FFFFFFu) | (2u << 30)};  // addr hi | type=2
  i32x8 g1 = {(int)(2u << 16),                      // data_size = 2 (4 bytes)
              (int)((nelem & 0xFFFFu) << 16),       // tensor_dim0[15:0]
              (int)((nelem >> 16) & 0xFFFFu),       // tensor_dim0[31:16], tensor_dim1 lo16=0 -> set below
              (int)((nelem & 0xFFFFu) << 16),       // tile_dim0 in bits 31:16
              0,                                    // tile_dim1=0 (1-D), tile_dim2=0
              (int)nelem,                           // tensor_dim0_stride[31:0]
              0, 0};
  g1[2] |= (1 << 16);                               // tensor_dim1 = 1
  i32x4 g2 = {0, 0, 0, 0};
  i32x4 g3 = {0, 0, 0, 0};
#if __clang_major__ >= 23
  i32x8 g4 = {0, 0, 0, 0, 0, 0, 0, 0};
  __builtin_amdgcn_tensor_load_to_lds(g0, g1, g2, g3, g4, 0);
#else
  __builtin_amdgcn_tensor_load_to_lds(g0, g1, g2, g3, 0);
#endif
}
#endif

// ---------------------------------------------------------------------------
// Generic C[M,N] = X[M,K] @ W[K,N], fp32 in/out, bf16 WMMA compute.
// One 16x16 tile per wave, 4 waves per block. Requires M%16==0, N%16==0, K%32==0.
// ---------------------------------------------------------------------------
__global__ __launch_bounds__(128) void gemm_wmma_bf16(
    const float* __restrict__ X, const float* __restrict__ W,
    float* __restrict__ C, int M, int N, int K) {
  int ntiles = N >> 4;
  int tile   = blockIdx.x * 4 + (threadIdx.x >> 5);
  int tm = tile / ntiles, tn = tile % ntiles;
  int lane = threadIdx.x & 31;
  int col  = lane & 15;
  int hv   = lane >> 4;
  int m = (tm << 4) + col;   // A row held by this lane
  int n = (tn << 4) + col;   // B/C column held by this lane
  v8f acc = {};
  for (int k0 = 0; k0 < K; k0 += 32) {
    if (k0 + 128 < K) {
      __builtin_prefetch(X + (size_t)m * K + k0 + 128, 0, 1);
      __builtin_prefetch(W + (size_t)(k0 + 128) * N + n, 0, 1);
    }
    v16bf a, b;
#pragma unroll
    for (int v = 0; v < 8; ++v) {
      int k = k0 + kpat(v, hv);
      float2 xa = *(const float2*)(X + (size_t)m * K + k);
      a[2 * v]     = f2bf(xa.x);
      a[2 * v + 1] = f2bf(xa.y);
      b[2 * v]     = f2bf(W[(size_t)k * N + n]);
      b[2 * v + 1] = f2bf(W[(size_t)(k + 1) * N + n]);
    }
    acc = __builtin_amdgcn_wmma_f32_16x16x32_bf16(false, a, false, b,
                                                  (short)0, acc, false, false);
  }
#pragma unroll
  for (int r = 0; r < 8; ++r) {
    int mm = (tm << 4) + r + 8 * hv;   // C layout: VGPR r -> row r (+8 upper half)
    C[(size_t)mm * N + n] = acc[r];
  }
}

// ---------------------------------------------------------------------------
// RoPE on q, in place. q layout: [b][s][qh][dh]. One thread per (pair).
// ---------------------------------------------------------------------------
__global__ __launch_bounds__(256) void rope_q_kernel(
    float* __restrict__ q, const int* __restrict__ positions) {
  int idx = blockIdx.x * blockDim.x + threadIdx.x;   // B*S*QH*(DH/2)
  int i  = idx & 31;
  int qh = (idx >> 5) & (QH_ - 1);
  int bs = idx >> 10;
  float pos = (float)positions[bs];
  float invf = __expf(-(2.0f * (float)i / (float)DH_) * 9.210340371976184f);
  float ang = pos * invf;
  float sn, cs;
  __sincosf(ang, &sn, &cs);
  float* p = q + ((size_t)(bs * QH_ + qh) << 6) + 2 * i;
  float e = p[0], o = p[1];
  p[0] = e * cs - o * sn;
  p[1] = o * cs + e * sn;
}

// ---------------------------------------------------------------------------
// k: RMS-norm over DH + k_scale + RoPE, in place. k layout: [b][s][h][dh].
// One thread per (b,s,h) row.
// ---------------------------------------------------------------------------
__global__ __launch_bounds__(256) void norm_rope_k_kernel(
    float* __restrict__ k, const int* __restrict__ positions,
    const float* __restrict__ kscale) {
  int idx = blockIdx.x * blockDim.x + threadIdx.x;   // B*S*H
  int h  = idx & (H_ - 1);
  int bs = idx >> 3;
  float* row = k + (size_t)(bs * H_ + h) * DH_;
  float ss = 0.f;
#pragma unroll 8
  for (int d = 0; d < DH_; ++d) { float t = row[d]; ss += t * t; }
  float inv_rms = rsqrtf(ss * (1.0f / DH_) + 1e-8f);
  float pos = (float)positions[bs];
#pragma unroll 4
  for (int i = 0; i < DH_ / 2; ++i) {
    float invf = __expf(-(2.0f * (float)i / (float)DH_) * 9.210340371976184f);
    float sn, cs;
    __sincosf(pos * invf, &sn, &cs);
    float e = row[2 * i]     * inv_rms * kscale[h * DH_ + 2 * i];
    float o = row[2 * i + 1] * inv_rms * kscale[h * DH_ + 2 * i + 1];
    row[2 * i]     = e * cs - o * sn;
    row[2 * i + 1] = o * cs + e * sn;
  }
}

// ---------------------------------------------------------------------------
// Flash-decoding attention. Block = (b,h); 4 waves, wave w = query group g=w.
// K/V tiles (32 keys x 64 fp32) staged in LDS: TDM DMA when the tile lies
// fully inside the cache (common case), cooperative vector loads otherwise.
// Online softmax; P re-laid-out via LDS for the P@V WMMAs.
// ---------------------------------------------------------------------------
__global__ __launch_bounds__(128) void attn_kernel(
    const float* __restrict__ q, const float* __restrict__ k_cache,
    const float* __restrict__ v_cache, const float* __restrict__ k_new,
    const float* __restrict__ v_new, const float* __restrict__ gate_logits,
    const int* __restrict__ cache_lengths, float* __restrict__ o) {
  __shared__ float  Ks[KT][DH_];
  __shared__ float  Vs[KT][DH_];
  __shared__ __bf16 Ps[4][16][KT];

  int b = blockIdx.x >> 3;
  int h = blockIdx.x & (H_ - 1);
  int tid  = threadIdx.x;
  int wave = tid >> 5;
  int lane = tid & 31;
  int col  = lane & 15;
  int hv   = lane >> 4;

  int kept = cache_lengths[b];
  if (kept > TOTAL_ - S_) kept = TOTAL_ - S_;

  int qh = h * G_ + wave;

  // Preload Q A-fragments once (A row = s = col; d per ISA K-pattern).
  v16bf aQ0, aQ1;
  {
    const float* qrow = q + ((size_t)((b * S_ + col) * QH_ + qh) << 6);
#pragma unroll
    for (int v = 0; v < 8; ++v) {
      int d = kpat(v, hv);
      float2 lo = *(const float2*)(qrow + d);
      float2 hi = *(const float2*)(qrow + d + 32);
      aQ0[2 * v] = f2bf(lo.x); aQ0[2 * v + 1] = f2bf(lo.y);
      aQ1[2 * v] = f2bf(hi.x); aQ1[2 * v + 1] = f2bf(hi.y);
    }
  }

  v8f accO0 = {}, accO1 = {}, accO2 = {}, accO3 = {};
  float runmax[8], runsum[8];
#pragma unroll
  for (int r = 0; r < 8; ++r) { runmax[r] = -3.0e38f; runsum[r] = 0.f; }

  const float* kc = k_cache + (size_t)(b * H_ + h) * CACHE_ * DH_;
  const float* vc = v_cache + (size_t)(b * H_ + h) * CACHE_ * DH_;

  for (int t = 0; t < TOTAL_ / KT; ++t) {
    __syncthreads();
#if HAVE_TDM
    if ((t + 1) * KT <= kept) {
      // Tile fully inside cached keys: descriptor-driven DMA (one wave issues).
      if (wave == 0) {
        tdm_load_1d((unsigned)(size_t)&Ks[0][0],
                    (unsigned long long)(kc + (size_t)t * KT * DH_), KT * DH_);
        tdm_load_1d((unsigned)(size_t)&Vs[0][0],
                    (unsigned long long)(vc + (size_t)t * KT * DH_), KT * DH_);
        __builtin_amdgcn_s_wait_tensorcnt(0);
      }
    } else
#endif
    {
      // Boundary / new-key / zero-pad tile: cooperative staging,
      // 128 threads x 4 float4 = 2048 floats per matrix.
#pragma unroll
      for (int i = 0; i < 4; ++i) {
        int f4  = (tid << 2) + i;        // 0..511
        int key = f4 >> 4;
        int d4  = (f4 & 15) << 2;
        int j   = t * KT + key;
        float4 kv = {0.f, 0.f, 0.f, 0.f}, vv = {0.f, 0.f, 0.f, 0.f};
        if (j < kept) {
          kv = *(const float4*)(kc + (size_t)j * DH_ + d4);
          vv = *(const float4*)(vc + (size_t)j * DH_ + d4);
        } else if (j < kept + S_) {
          int ssrc = j - kept;
          kv = *(const float4*)(k_new + (size_t)((b * S_ + ssrc) * H_ + h) * DH_ + d4);
          vv = *(const float4*)(v_new + (size_t)((b * S_ + ssrc) * H_ + h) * DH_ + d4);
        }
        *(float4*)&Ks[key][d4] = kv;
        *(float4*)&Vs[key][d4] = vv;
      }
    }
    __syncthreads();

    // Scores: 16 q rows x 32 keys, as two 16x16 C tiles; K-dim 64 = 2 WMMAs each.
    v8f sc0 = {}, sc1 = {};
    {
      v16bf b0, b1;
#pragma unroll
      for (int v = 0; v < 8; ++v) {
        int d = kpat(v, hv);
        float2 lo = *(const float2*)&Ks[col][d];
        float2 hi = *(const float2*)&Ks[col][d + 32];
        b0[2 * v] = f2bf(lo.x); b0[2 * v + 1] = f2bf(lo.y);
        b1[2 * v] = f2bf(hi.x); b1[2 * v + 1] = f2bf(hi.y);
      }
      sc0 = __builtin_amdgcn_wmma_f32_16x16x32_bf16(false, aQ0, false, b0, (short)0, sc0, false, false);
      sc0 = __builtin_amdgcn_wmma_f32_16x16x32_bf16(false, aQ1, false, b1, (short)0, sc0, false, false);
#pragma unroll
      for (int v = 0; v < 8; ++v) {
        int d = kpat(v, hv);
        float2 lo = *(const float2*)&Ks[16 + col][d];
        float2 hi = *(const float2*)&Ks[16 + col][d + 32];
        b0[2 * v] = f2bf(lo.x); b0[2 * v + 1] = f2bf(lo.y);
        b1[2 * v] = f2bf(hi.x); b1[2 * v + 1] = f2bf(hi.y);
      }
      sc1 = __builtin_amdgcn_wmma_f32_16x16x32_bf16(false, aQ0, false, b0, (short)0, sc1, false, false);
      sc1 = __builtin_amdgcn_wmma_f32_16x16x32_bf16(false, aQ1, false, b1, (short)0, sc1, false, false);
    }

    // Online softmax per row (C layout: reg r -> row r + 8*hv; cols across 16 lanes).
#pragma unroll
    for (int r = 0; r < 8; ++r) {
      float s0 = sc0[r] * 0.125f;   // 1/sqrt(DH)
      float s1 = sc1[r] * 0.125f;
      float tmax = fmaxf(s0, s1);
#pragma unroll
      for (int mk = 1; mk < 16; mk <<= 1)
        tmax = fmaxf(tmax, __shfl_xor(tmax, mk, 16));
      float newmax = fmaxf(runmax[r], tmax);
      float corr = __expf(runmax[r] - newmax);
      runmax[r] = newmax;
      float p0 = __expf(s0 - newmax);
      float p1 = __expf(s1 - newmax);
      float psum = p0 + p1;
#pragma unroll
      for (int mk = 1; mk < 16; mk <<= 1)
        psum += __shfl_xor(psum, mk, 16);
      runsum[r] = runsum[r] * corr + psum;
      accO0[r] *= corr; accO1[r] *= corr; accO2[r] *= corr; accO3[r] *= corr;
      int rowm = r + 8 * hv;
      Ps[wave][rowm][col]      = f2bf(p0);   // per-wave region: no __syncthreads needed
      Ps[wave][rowm][16 + col] = f2bf(p1);
    }
    // Same-wave LDS write -> cross-lane read: force ordering.
    asm volatile("s_wait_dscnt 0" ::: "memory");

    // P(16x32) @ V(32x64): A-frag from Ps, B-frags from Vs, 4 column tiles.
    v16bf aP;
#pragma unroll
    for (int v = 0; v < 8; ++v) {
      int kk = kpat(v, hv);
      aP[2 * v]     = Ps[wave][col][kk];
      aP[2 * v + 1] = Ps[wave][col][kk + 1];
    }
#pragma unroll
    for (int nt = 0; nt < 4; ++nt) {
      v16bf bV;
#pragma unroll
      for (int v = 0; v < 8; ++v) {
        int kk = kpat(v, hv);
        bV[2 * v]     = f2bf(Vs[kk][nt * 16 + col]);
        bV[2 * v + 1] = f2bf(Vs[kk + 1][nt * 16 + col]);
      }
      if      (nt == 0) accO0 = __builtin_amdgcn_wmma_f32_16x16x32_bf16(false, aP, false, bV, (short)0, accO0, false, false);
      else if (nt == 1) accO1 = __builtin_amdgcn_wmma_f32_16x16x32_bf16(false, aP, false, bV, (short)0, accO1, false, false);
      else if (nt == 2) accO2 = __builtin_amdgcn_wmma_f32_16x16x32_bf16(false, aP, false, bV, (short)0, accO2, false, false);
      else              accO3 = __builtin_amdgcn_wmma_f32_16x16x32_bf16(false, aP, false, bV, (short)0, accO3, false, false);
    }
  }

  // Epilogue: normalize, gate, store o[b][s][qh][dh].
#pragma unroll
  for (int r = 0; r < 8; ++r) {
    int rowm = r + 8 * hv;   // = s
    float gl = gate_logits[(b * S_ + rowm) * QH_ + qh];
    float gate = 1.0f / (1.0f + __expf(-gl));
    float sc = gate / runsum[r];
    float* orow = o + (size_t)((b * S_ + rowm) * QH_ + qh) * DH_;
    orow[0 * 16 + col] = accO0[r] * sc;
    orow[1 * 16 + col] = accO1[r] * sc;
    orow[2 * 16 + col] = accO2[r] * sc;
    orow[3 * 16 + col] = accO3[r] * sc;
  }
}

// ---------------------------------------------------------------------------
extern "C" void kernel_launch(void* const* d_in, const int* in_sizes, int n_in,
                              void* d_out, int out_size, void* d_ws, size_t ws_size,
                              hipStream_t stream) {
  (void)in_sizes; (void)n_in; (void)out_size; (void)ws_size;
  const float* x             = (const float*)d_in[0];
  const int*   positions     = (const int*)d_in[1];
  const int*   cache_lengths = (const int*)d_in[2];
  const float* k_cache       = (const float*)d_in[3];
  const float* v_cache       = (const float*)d_in[4];
  const float* Wq            = (const float*)d_in[5];
  const float* Wk            = (const float*)d_in[6];
  const float* Wv            = (const float*)d_in[7];
  const float* Wo            = (const float*)d_in[8];
  const float* Wg            = (const float*)d_in[9];
  const float* k_scale       = (const float*)d_in[10];

  float* ws   = (float*)d_ws;
  float* q_ws = ws;                                   // 512*2048
  float* k_ws = q_ws + (size_t)512 * 2048;            // 512*512
  float* v_ws = k_ws + (size_t)512 * 512;             // 512*512
  float* g_ws = v_ws + (size_t)512 * 512;             // 512*32
  float* o_ws = g_ws + (size_t)512 * 32;              // 512*2048

  const int M = 512;  // B*S
  gemm_wmma_bf16<<<dim3((M / 16) * (2048 / 16) / 4), 128, 0, stream>>>(x, Wq, q_ws, M, 2048, 2048);
  gemm_wmma_bf16<<<dim3((M / 16) * (512 / 16) / 4),  128, 0, stream>>>(x, Wk, k_ws, M, 512, 2048);
  gemm_wmma_bf16<<<dim3((M / 16) * (512 / 16) / 4),  128, 0, stream>>>(x, Wv, v_ws, M, 512, 2048);
  gemm_wmma_bf16<<<dim3((M / 16) * (32 / 16) / 4),   128, 0, stream>>>(x, Wg, g_ws, M, 32, 2048);

  rope_q_kernel<<<dim3((512 * 32 * 32) / 256), 256, 0, stream>>>(q_ws, positions);
  norm_rope_k_kernel<<<dim3(4096 / 256), 256, 0, stream>>>(k_ws, positions, k_scale);

  attn_kernel<<<dim3(B_ * H_), 128, 0, stream>>>(q_ws, k_cache, v_cache, k_ws, v_ws,
                                                 g_ws, cache_lengths, o_ws);

  gemm_wmma_bf16<<<dim3((M / 16) * (2048 / 16) / 4), 128, 0, stream>>>(o_ws, Wo, (float*)d_out, M, 2048, 2048);
}
